// MultiHeadAttention_42348377539173
// MI455X (gfx1250) — matrix operand order repaired and checked by reading, not compiled, not run on hardware
//
#include <hip/hip_runtime.h>
#include <hip/hip_bf16.h>

// ---------------------------------------------------------------------------
// MHA with QK-norm + additive bias, CDNA5 (gfx1250) WMMA bf16 implementation.
//   S=2048, d_model=1024, H=16, d_k=64
//   - bf16 WMMA (v_wmma_f32_16x16x32_bf16) for all GEMMs + attention
//   - flash-style online softmax (bias streamed once, prefetched)
//   - async global->LDS staging (ASYNCcnt) where no data transform is needed
// ---------------------------------------------------------------------------

typedef unsigned short u16;
typedef __attribute__((ext_vector_type(16))) __bf16 bf16x16;
typedef __attribute__((ext_vector_type(8)))  float  f32x8;
typedef int v4i __attribute__((vector_size(16)));   // param type of async builtin

#define SEQ     2048
#define DMODEL  1024
#define NHEAD   16
#define DK      64
#define ATT_SCALE 0.125f   // 1/sqrt(64)

#if defined(__has_builtin)
#  if __has_builtin(__builtin_amdgcn_global_load_async_to_lds_b128)
#    define USE_ASYNC_LDS 1
#  endif
#endif

// 16-byte global -> LDS copy; async DMA path on CDNA5 when available.
__device__ __forceinline__ void copy_b128_to_lds(void* lds, const void* g) {
#ifdef USE_ASYNC_LDS
    __builtin_amdgcn_global_load_async_to_lds_b128(
        (v4i*)g, (v4i*)lds, /*offset=*/0, /*cpol=*/0);
#else
    *(uint4*)lds = *(const uint4*)g;
#endif
}

__device__ __forceinline__ void async_wait_all() {
#ifdef USE_ASYNC_LDS
#  if __has_builtin(__builtin_amdgcn_s_wait_asynccnt)
    __builtin_amdgcn_s_wait_asynccnt(0);
#  else
    asm volatile("s_wait_asynccnt 0" ::: "memory");
#  endif
#endif
}

// float -> bf16 (round to nearest even)
__device__ __forceinline__ u16 f2bf(float x) {
    unsigned int u = __float_as_uint(x);
    unsigned int r = u + 0x7FFFu + ((u >> 16) & 1u);
    return (u16)(r >> 16);
}

// Load a 16-element bf16 WMMA fragment from two 16-byte LDS chunks.
__device__ __forceinline__ bf16x16 ld_frag(const u16* p0, const u16* p1) {
    union { bf16x16 v; uint4 u[2]; } f;
    f.u[0] = *(const uint4*)p0;
    f.u[1] = *(const uint4*)p1;
    return f.v;
}

__device__ __forceinline__ f32x8 wmma_bf16(bf16x16 a, bf16x16 b, f32x8 c) {
    return __builtin_amdgcn_wmma_f32_16x16x32_bf16(
        /*neg_a=*/false, a, /*neg_b=*/false, b,
        /*c_mod=*/(short)0, c, /*reuse_a=*/false, /*reuse_b=*/false);
}

// ===========================================================================
// Kernel 1: fused QKV projections + per-head L2 norm (Q,K) -> bf16 [H][S][64]
//   grid = (16 m-tiles, 16 heads, 3 {Q,K,V}),  block = 256 (8 waves)
//   tile: 128(M) x 64(N=one head) x 32(K-step)
// ===========================================================================
__global__ __launch_bounds__(256)
void proj_kernel(const float* __restrict__ Ain0, const float* __restrict__ Ain1,
                 const float* __restrict__ Ain2,
                 const float* __restrict__ Wq,  const float* __restrict__ Wk,
                 const float* __restrict__ Wv,
                 u16* __restrict__ Qn, u16* __restrict__ Kn, u16* __restrict__ Vn)
{
    __shared__ __align__(16) u16  As[128][40];   // [m][k]
    __shared__ __align__(16) u16  Bs[64][40];    // [n][k]  (W transposed)
    __shared__ float Cs[128][65];

    const int z = blockIdx.z;
    const float* A = (z == 0) ? Ain0 : (z == 1) ? Ain1 : Ain2;
    const float* W = (z == 0) ? Wq   : (z == 1) ? Wk   : Wv;
    u16*       Out = (z == 0) ? Qn   : (z == 1) ? Kn   : Vn;
    const bool norm = (z < 2);

    const int m0 = blockIdx.x * 128;
    const int h  = blockIdx.y;
    const int n0 = h * 64;
    const int t  = threadIdx.x;
    const int lane = t & 31, wave = t >> 5;

    f32x8 acc[4];
    for (int j = 0; j < 4; ++j)
        for (int e = 0; e < 8; ++e) acc[j][e] = 0.0f;

    const int mrow = wave * 16 + (lane & 15);
    const int kgA  = (lane < 16) ? 0 : 8;
    const int kgB  = (lane < 16) ? 0 : 8;

    for (int k0 = 0; k0 < DMODEL; k0 += 32) {
        __syncthreads();
        // stage A tile 128x32 (fp32 -> bf16, VALU convert)
        #pragma unroll
        for (int i = 0; i < 4; ++i) {
            int idx = t + i * 256;            // 1024 float4 units
            int row = idx >> 3, c4 = (idx & 7) * 4;
            float4 f = *(const float4*)(A + (size_t)(m0 + row) * DMODEL + k0 + c4);
            u16* d = &As[row][c4];
            d[0] = f2bf(f.x); d[1] = f2bf(f.y); d[2] = f2bf(f.z); d[3] = f2bf(f.w);
        }
        // stage W tile 32x64 transposed -> Bs[n][k]
        #pragma unroll
        for (int i = 0; i < 2; ++i) {
            int idx = t + i * 256;            // 512 float4 units
            int kr = idx >> 4, c4 = (idx & 15) * 4;
            float4 f = *(const float4*)(W + (size_t)(k0 + kr) * DMODEL + n0 + c4);
            Bs[c4 + 0][kr] = f2bf(f.x); Bs[c4 + 1][kr] = f2bf(f.y);
            Bs[c4 + 2][kr] = f2bf(f.z); Bs[c4 + 3][kr] = f2bf(f.w);
        }
        __syncthreads();

        bf16x16 afrag = ld_frag(&As[mrow][kgA], &As[mrow][16 + kgA]);
        #pragma unroll
        for (int j = 0; j < 4; ++j) {
            int n = j * 16 + (lane & 15);
            bf16x16 bfrag = ld_frag(&Bs[n][kgB], &Bs[n][16 + kgB]);
            acc[j] = wmma_bf16(afrag, bfrag, acc[j]);
        }
    }

    __syncthreads();
    // C layout: lane = col within 16-tile, vgpr r = row (0..7 / 8..15 by lane half)
    #pragma unroll
    for (int j = 0; j < 4; ++j)
        #pragma unroll
        for (int r = 0; r < 8; ++r)
            Cs[wave * 16 + (lane >> 4) * 8 + r][j * 16 + (lane & 15)] = acc[j][r];
    __syncthreads();

    if (t < 128) {
        float inv = 1.0f;
        if (norm) {
            float s = 0.0f;
            for (int c = 0; c < 64; ++c) { float v = Cs[t][c]; s += v * v; }
            inv = 1.0f / (sqrtf(s) + 1e-6f);
        }
        u16* outRow = Out + ((size_t)h * SEQ + (m0 + t)) * DK;
        for (int c = 0; c < 64; c += 2) {
            unsigned int pack = (unsigned int)f2bf(Cs[t][c] * inv) |
                                ((unsigned int)f2bf(Cs[t][c + 1] * inv) << 16);
            *(unsigned int*)(outRow + c) = pack;
        }
    }
}

// ===========================================================================
// Kernel 2: flash-style attention per head with additive bias.
//   grid = (16 q-tiles, 16 heads), block = 256 (8 waves, 16 queries/wave)
// ===========================================================================
__global__ __launch_bounds__(256)
void attn_kernel(const u16* __restrict__ Qn, const u16* __restrict__ Kn,
                 const u16* __restrict__ Vn, const float* __restrict__ bias,
                 u16* __restrict__ attended)
{
    __shared__ __align__(16) u16 Qs[128][72];     // [q][d]
    __shared__ __align__(16) u16 Ks[64][72];      // [key][d]  == B layout for Q*K^T
    __shared__ __align__(16) u16 Vt[64][72];      // [d][key]  == B layout for P*V
    __shared__ __align__(16) u16 Ps[8][16][72];   // per-wave P tile [q][key]

    const int h  = blockIdx.y;
    const int q0 = blockIdx.x * 128;
    const int t  = threadIdx.x;
    const int lane = t & 31, wave = t >> 5;

    // stage Q tile once (async bf16 copy, 16B units)
    #pragma unroll
    for (int i = 0; i < 4; ++i) {
        int idx = t + i * 256;                 // 1024 uint4 units (8 bf16 each)
        int row = idx >> 3, c = (idx & 7) * 8;
        copy_b128_to_lds(&Qs[row][c],
                         Qn + ((size_t)h * SEQ + q0 + row) * DK + c);
    }
    // prefetch first bias tile while Q streams in
    if (t < 128)
        __builtin_prefetch(bias + ((size_t)h * SEQ + q0 + t) * SEQ, 0, 0);

    f32x8 accO[4];
    for (int j = 0; j < 4; ++j)
        for (int e = 0; e < 8; ++e) accO[j][e] = 0.0f;
    float mrun[8], lrun[8];
    #pragma unroll
    for (int r = 0; r < 8; ++r) { mrun[r] = -3.0e38f; lrun[r] = 0.0f; }

    const int mrow  = wave * 16 + (lane & 15);
    const int rbase = (lane >> 4) * 8;
    const int kgA   = (lane < 16) ? 0 : 8;
    const int kgB   = (lane < 16) ? 0 : 8;

    for (int kt = 0; kt < SEQ; kt += 64) {
        __syncthreads();
        // stage K tile 64x64 (natural layout, async 16B copies)
        #pragma unroll
        for (int i = 0; i < 2; ++i) {
            int idx = t + i * 256;             // 512 uint4 units
            int row = idx >> 3, c = (idx & 7) * 8;
            copy_b128_to_lds(&Ks[row][c],
                             Kn + ((size_t)h * SEQ + kt + row) * DK + c);
        }
        // stage V tile transposed: Vt[d][key] (needs transform -> VALU path)
        {
            int key = t >> 2, dbase = (t & 3) * 16;
            const u16* src = Vn + ((size_t)h * SEQ + kt + key) * DK + dbase;
            #pragma unroll
            for (int i = 0; i < 16; ++i) Vt[dbase + i][key] = src[i];
        }
        // prefetch next key-tile's bias rows while we compute this one
        if (kt + 64 < SEQ && t < 128)
            __builtin_prefetch(bias + ((size_t)h * SEQ + q0 + t) * SEQ + (kt + 64),
                               0, 0);
        async_wait_all();
        __syncthreads();

        // ---- S = Q * K^T (8 WMMA per wave) ----
        f32x8 s[4];
        for (int j = 0; j < 4; ++j)
            for (int e = 0; e < 8; ++e) s[j][e] = 0.0f;
        #pragma unroll
        for (int st = 0; st < 2; ++st) {
            bf16x16 a = ld_frag(&Qs[mrow][st * 32 + kgA], &Qs[mrow][st * 32 + 16 + kgA]);
            #pragma unroll
            for (int j = 0; j < 4; ++j) {
                int n = j * 16 + (lane & 15);
                bf16x16 b = ld_frag(&Ks[n][st * 32 + kgB], &Ks[n][st * 32 + 16 + kgB]);
                s[j] = wmma_bf16(a, b, s[j]);
            }
        }

        // ---- bias + online softmax (per row; 16-lane shuffle reductions) ----
        #pragma unroll
        for (int r = 0; r < 8; ++r) {
            int qg = q0 + wave * 16 + rbase + r;
            float v[4], rmax = -3.0e38f;
            #pragma unroll
            for (int j = 0; j < 4; ++j) {
                int kg = kt + j * 16 + (lane & 15);
                float b = bias[((size_t)h * SEQ + qg) * SEQ + kg];
                v[j] = s[j][r] * ATT_SCALE + b;
                rmax = fmaxf(rmax, v[j]);
            }
            for (int m = 1; m < 16; m <<= 1) rmax = fmaxf(rmax, __shfl_xor(rmax, m));
            float mnew  = fmaxf(mrun[r], rmax);
            float alpha = __expf(mrun[r] - mnew);
            float rsum = 0.0f;
            #pragma unroll
            for (int j = 0; j < 4; ++j) {
                float p = __expf(v[j] - mnew);
                rsum += p;
                Ps[wave][rbase + r][j * 16 + (lane & 15)] = f2bf(p);
            }
            for (int m = 1; m < 16; m <<= 1) rsum += __shfl_xor(rsum, m);
            lrun[r] = lrun[r] * alpha + rsum;
            mrun[r] = mnew;
            #pragma unroll
            for (int j = 0; j < 4; ++j) accO[j][r] *= alpha;
        }
        __syncthreads();

        // ---- O += P * V (8 WMMA per wave) ----
        #pragma unroll
        for (int st = 0; st < 2; ++st) {
            bf16x16 a = ld_frag(&Ps[wave][lane & 15][st * 32 + kgA],
                                &Ps[wave][lane & 15][st * 32 + 16 + kgA]);
            #pragma unroll
            for (int j = 0; j < 4; ++j) {
                int n = j * 16 + (lane & 15);
                bf16x16 b = ld_frag(&Vt[n][st * 32 + kgB], &Vt[n][st * 32 + 16 + kgB]);
                accO[j] = wmma_bf16(a, b, accO[j]);
            }
        }
    }

    // finalize: attended[q][h*64 + d] bf16
    #pragma unroll
    for (int r = 0; r < 8; ++r) {
        int qg = q0 + wave * 16 + rbase + r;
        float invl = 1.0f / lrun[r];
        #pragma unroll
        for (int j = 0; j < 4; ++j) {
            int col = h * DK + j * 16 + (lane & 15);
            attended[(size_t)qg * DMODEL + col] = f2bf(accO[j][r] * invl);
        }
    }
}

// ===========================================================================
// Kernel 3: output projection  out = attended(bf16) @ W_o  -> fp32
//   grid = (16, 16), block = 256; tile 128x64x32
// ===========================================================================
__global__ __launch_bounds__(256)
void oproj_kernel(const u16* __restrict__ Aus, const float* __restrict__ W,
                  float* __restrict__ Out)
{
    __shared__ __align__(16) u16  As[128][40];
    __shared__ __align__(16) u16  Bs[64][40];
    __shared__ float Cs[128][65];

    const int m0 = blockIdx.x * 128;
    const int n0 = blockIdx.y * 64;
    const int t  = threadIdx.x;
    const int lane = t & 31, wave = t >> 5;

    f32x8 acc[4];
    for (int j = 0; j < 4; ++j)
        for (int e = 0; e < 8; ++e) acc[j][e] = 0.0f;

    const int mrow = wave * 16 + (lane & 15);
    const int kgA  = (lane < 16) ? 0 : 8;
    const int kgB  = (lane < 16) ? 0 : 8;

    for (int k0 = 0; k0 < DMODEL; k0 += 32) {
        __syncthreads();
        // stage A tile 128x32 bf16 (async 16B copies: 512 units, 2 per thread)
        #pragma unroll
        for (int i = 0; i < 2; ++i) {
            int idx = t + i * 256;
            int row = idx >> 2, c8 = (idx & 3) * 8;
            copy_b128_to_lds(&As[row][c8],
                             Aus + (size_t)(m0 + row) * DMODEL + k0 + c8);
        }
        // stage W tile 32x64 transposed (fp32 -> bf16, VALU)
        #pragma unroll
        for (int i = 0; i < 2; ++i) {
            int idx = t + i * 256;
            int kr = idx >> 4, c4 = (idx & 15) * 4;
            float4 f = *(const float4*)(W + (size_t)(k0 + kr) * DMODEL + n0 + c4);
            Bs[c4 + 0][kr] = f2bf(f.x); Bs[c4 + 1][kr] = f2bf(f.y);
            Bs[c4 + 2][kr] = f2bf(f.z); Bs[c4 + 3][kr] = f2bf(f.w);
        }
        async_wait_all();
        __syncthreads();

        bf16x16 afrag = ld_frag(&As[mrow][kgA], &As[mrow][16 + kgA]);
        #pragma unroll
        for (int j = 0; j < 4; ++j) {
            int n = j * 16 + (lane & 15);
            bf16x16 bfrag = ld_frag(&Bs[n][kgB], &Bs[n][16 + kgB]);
            acc[j] = wmma_bf16(afrag, bfrag, acc[j]);
        }
    }

    __syncthreads();
    #pragma unroll
    for (int j = 0; j < 4; ++j)
        #pragma unroll
        for (int r = 0; r < 8; ++r)
            Cs[wave * 16 + (lane >> 4) * 8 + r][j * 16 + (lane & 15)] = acc[j][r];
    __syncthreads();

    if (t < 128) {
        float* o = Out + (size_t)(m0 + t) * DMODEL + n0;
        for (int c = 0; c < 64; c += 4) {
            float4 f;
            f.x = Cs[t][c]; f.y = Cs[t][c + 1]; f.z = Cs[t][c + 2]; f.w = Cs[t][c + 3];
            *(float4*)(o + c) = f;
        }
    }
}

// ===========================================================================
extern "C" void kernel_launch(void* const* d_in, const int* in_sizes, int n_in,
                              void* d_out, int out_size, void* d_ws, size_t ws_size,
                              hipStream_t stream)
{
    const float* Qin  = (const float*)d_in[0];
    const float* Kin  = (const float*)d_in[1];
    const float* Vin  = (const float*)d_in[2];
    const float* bias = (const float*)d_in[3];
    const float* Wq   = (const float*)d_in[4];
    const float* Wk   = (const float*)d_in[5];
    const float* Wv   = (const float*)d_in[6];
    const float* Wo   = (const float*)d_in[7];

    const size_t per = (size_t)NHEAD * SEQ * DK;   // elements per head-major buffer
    u16* Qn  = (u16*)d_ws;
    u16* Kn  = Qn + per;
    u16* Vn  = Kn + per;
    u16* att = Vn + per;                            // [SEQ][DMODEL] bf16

    proj_kernel<<<dim3(16, 16, 3), 256, 0, stream>>>(Qin, Kin, Vin, Wq, Wk, Wv,
                                                     Qn, Kn, Vn);
    attn_kernel<<<dim3(16, 16), 256, 0, stream>>>(Qn, Kn, Vn, bias, att);
    oproj_kernel<<<dim3(16, 16), 256, 0, stream>>>(att, Wo, (float*)d_out);
}
